// ExphormerModel_438086664593
// MI455X (gfx1250) — matrix operand order repaired and checked by reading, not compiled, not run on hardware
//
#include <hip/hip_runtime.h>
#include <hip/hip_bf16.h>

// ---------------- types ----------------
typedef __bf16 bf16t;
typedef __attribute__((ext_vector_type(16))) __bf16 v16bf;
typedef __attribute__((ext_vector_type(8)))  float  v8f;
typedef __attribute__((ext_vector_type(4)))  unsigned int u32x4;
typedef __attribute__((ext_vector_type(4)))  float  f32x4;
typedef __attribute__((__vector_size__(16))) int v4i_vs;   // matches builtin param type

#define NROWS 50000
#define DMODEL 128
#define NHEAD 8
#define EDGES 800000
#define DFF2 256

#ifndef __has_builtin
#define __has_builtin(x) 0
#endif

#if __has_builtin(__builtin_amdgcn_global_load_async_to_lds_b128)
#define EDGE_ASYNC 1
#else
#define EDGE_ASYNC 0
#endif

#define AS1 __attribute__((address_space(1)))
#define AS3 __attribute__((address_space(3)))

union Frag { v16bf v; u32x4 u[2]; bf16t b[16]; };

// A-operand: lane L (q=L/16, r=L%16) holds row m=r, k = base+8q+{0..7} and base+8q+16+{0..7}
__device__ inline v16bf load_afrag(const bf16t* p) {
  Frag f;
  f.u[0] = *(const u32x4*)(p);
  f.u[1] = *(const u32x4*)(p + 16);
  return f.v;
}
// B-operand (pre-packed): lane L reads 16 contiguous bf16 at tile_base + L*16
__device__ inline v16bf load_bfrag(const bf16t* p) {
  Frag f;
  f.u[0] = *(const u32x4*)(p);
  f.u[1] = *(const u32x4*)(p + 8);
  return f.v;
}
// Assemble A-fragment directly from an f32 row in LDS (converts f32->bf16).
__device__ inline v16bf make_afrag_f32(const float* row) {
  f32x4 c0 = *(const f32x4*)(row);
  f32x4 c1 = *(const f32x4*)(row + 4);
  f32x4 c2 = *(const f32x4*)(row + 16);
  f32x4 c3 = *(const f32x4*)(row + 20);
  Frag f;
  f.b[0]  = (bf16t)c0.x; f.b[1]  = (bf16t)c0.y; f.b[2]  = (bf16t)c0.z; f.b[3]  = (bf16t)c0.w;
  f.b[4]  = (bf16t)c1.x; f.b[5]  = (bf16t)c1.y; f.b[6]  = (bf16t)c1.z; f.b[7]  = (bf16t)c1.w;
  f.b[8]  = (bf16t)c2.x; f.b[9]  = (bf16t)c2.y; f.b[10] = (bf16t)c2.z; f.b[11] = (bf16t)c2.w;
  f.b[12] = (bf16t)c3.x; f.b[13] = (bf16t)c3.y; f.b[14] = (bf16t)c3.z; f.b[15] = (bf16t)c3.w;
  return f.v;
}
__device__ inline v8f wmma_bf16(v16bf a, v16bf b, v8f c) {
  return __builtin_amdgcn_wmma_f32_16x16x32_bf16(false, a, false, b, (short)0, c, false, false);
}

// ---------------- pre-pass kernels ----------------
// Pack f32 weight [K, Nsrc] row-major into bf16 WMMA B-operand tiles.
// Tile (kt, nt) occupies 512 elems; within tile, lane L slot j:
//   k = kt*32 + (L/16)*8 + (j%8) + (j/8)*16, n = nt*16 + (L%16)   (n >= Nsrc -> 0)
__global__ void pack_w(const float* __restrict__ W, bf16t* __restrict__ out,
                       int KT, int NT, int Nsrc) {
  int tid = blockIdx.x * 256 + threadIdx.x;
  int total = KT * NT * 512;
  if (tid >= total) return;
  int tile = tid >> 9, within = tid & 511;
  int L = within >> 4, j = within & 15;
  int nt = tile % NT, kt = tile / NT;
  int k = kt * 32 + (L >> 4) * 8 + (j & 7) + (j >> 3) * 16;
  int n = nt * 16 + (L & 15);
  out[tid] = (n < Nsrc) ? (bf16t)W[k * Nsrc + n] : (bf16t)0.0f;
}

// Pack WQ|WK|WV concatenated to Nout=384 (24 n-tiles)
__global__ void pack_qkv(const float* __restrict__ WQ, const float* __restrict__ WK,
                         const float* __restrict__ WV, bf16t* __restrict__ out) {
  int tid = blockIdx.x * 256 + threadIdx.x;
  if (tid >= 4 * 24 * 512) return;
  int tile = tid >> 9, within = tid & 511;
  int L = within >> 4, j = within & 15;
  int nt = tile % 24, kt = tile / 24;
  int k = kt * 32 + (L >> 4) * 8 + (j & 7) + (j >> 3) * 16;
  int n = nt * 16 + (L & 15);
  const float* W = (n < 128) ? WQ : (n < 256) ? WK : WV;
  out[tid] = (bf16t)W[k * 128 + (n & 127)];
}

__global__ void f32_to_bf16(const float* __restrict__ in, bf16t* __restrict__ out, int n) {
  int i = blockIdx.x * 256 + threadIdx.x;
  if (i < n) out[i] = (bf16t)in[i];
}

__global__ void zero_f32(float* __restrict__ p, int n) {
  int i = blockIdx.x * 256 + threadIdx.x;
  if (i < n) p[i] = 0.0f;
}

// ---------------- QKV projection: QKV[N,384] = h @ [WQ WK WV] ----------------
__global__ void qkv_gemm(const bf16t* __restrict__ hb, const bf16t* __restrict__ Wp,
                         float* __restrict__ QKV) {
  int rt = blockIdx.x;          // 3125 row tiles
  int part = blockIdx.y;        // 0=Q, 1=K, 2=V (8 n-tiles each)
  int L = threadIdx.x, q = L >> 4, r = L & 15;
  const bf16t* Arow = hb + (long)(rt * 16 + r) * 128;
  v16bf a[4];
#pragma unroll
  for (int s = 0; s < 4; ++s) a[s] = load_afrag(Arow + 32 * s + 8 * q);
#pragma unroll
  for (int c = 0; c < 8; ++c) {
    int nt = part * 8 + c;
    v8f acc = {};
#pragma unroll
    for (int s = 0; s < 4; ++s)
      acc = wmma_bf16(a[s], load_bfrag(Wp + ((s * 24 + nt) << 9) + L * 16), acc);
    int n = nt * 16 + r;
#pragma unroll
    for (int i = 0; i < 8; ++i)
      QKV[(long)(rt * 16 + i + 8 * q) * 384 + n] = acc[i];
  }
}

// ---------------- fused edge kernel: E-GEMM + score + exp + scatter ----------------
__global__ void edge_kernel(const float* __restrict__ edge_attr,
                            const long long* __restrict__ edge_index,
                            const float* __restrict__ QKV,
                            const bf16t* __restrict__ WEp, const bf16t* __restrict__ WEbp,
                            const float* __restrict__ bEb,
                            float* __restrict__ wV, float* __restrict__ Zacc) {
  __shared__ float sBias[16][8];
  __shared__ int sSrc[16], sDst[16];
  int e0 = blockIdx.x * 16;
  int L = threadIdx.x, q = L >> 4, r = L & 15;

  if (L < 16) sSrc[L] = (int)edge_index[e0 + L];
  else        sDst[L - 16] = (int)edge_index[(long)EDGES + e0 + (L - 16)];

  v16bf a[4];

#if EDGE_ASYNC
  // gfx1250 async path: DMA the 16x128 f32 tile straight into LDS
  // (GLOBAL_LOAD_ASYNC_TO_LDS_B128, ASYNCcnt-tracked), then build bf16
  // A-fragments from LDS with packed converts. No VGPR round-trip.
  __shared__ __align__(16) float sF[16 * 128];
#pragma unroll
  for (int it = 0; it < 16; ++it) {
    const float* g = edge_attr + (long)(e0 + it) * 128 + L * 4;
    float* l = sF + it * 128 + L * 4;
    __builtin_amdgcn_global_load_async_to_lds_b128(
        (AS1 v4i_vs*)(uintptr_t)g,
        (AS3 v4i_vs*)(unsigned)(uintptr_t)l,
        0, 0);
  }
#if __has_builtin(__builtin_amdgcn_s_wait_asynccnt)
  __builtin_amdgcn_s_wait_asynccnt(0);
#else
  asm volatile("s_wait_asynccnt 0" ::: "memory");
#endif
  __syncthreads();
#pragma unroll
  for (int s = 0; s < 4; ++s)
    a[s] = make_afrag_f32(sF + r * 128 + 32 * s + 8 * q);
#else
  // fallback: stage via VGPRs with non-temporal loads (protect L2 residency
  // of QKV/wV from the 410 MB edge_attr stream), convert, park bf16 in LDS.
  __shared__ __align__(16) bf16t sA[16 * 128];
#pragma unroll
  for (int it = 0; it < 16; ++it) {
    f32x4 v = __builtin_nontemporal_load((const f32x4*)(edge_attr + (long)(e0 + it) * 128 + L * 4));
    bf16t* d = sA + it * 128 + L * 4;
    d[0] = (bf16t)v.x; d[1] = (bf16t)v.y; d[2] = (bf16t)v.z; d[3] = (bf16t)v.w;
  }
  __syncthreads();
#pragma unroll
  for (int s = 0; s < 4; ++s) a[s] = load_afrag(sA + r * 128 + 32 * s + 8 * q);
#endif

  // E_bias = attr @ WEb (padded to 16 cols) + bEb -> LDS
  v8f accB = {};
#pragma unroll
  for (int s = 0; s < 4; ++s)
    accB = wmma_bf16(a[s], load_bfrag(WEbp + (s << 9) + L * 16), accB);
  if (r < 8) {
    float bb = bEb[r];
#pragma unroll
    for (int i = 0; i < 8; ++i) sBias[i + 8 * q][r] = accB[i] + bb;
  }
  __syncthreads();

  // per head: E column-tile == head (DH==16). acc lane holds (m=i+8q, dh=r).
#pragma unroll
  for (int h = 0; h < 8; ++h) {
    v8f accE = {};
#pragma unroll
    for (int s = 0; s < 4; ++s)
      accE = wmma_bf16(a[s], load_bfrag(WEp + ((s * 8 + h) << 9) + L * 16), accE);
    int col = h * 16 + r;
#pragma unroll
    for (int i = 0; i < 8; ++i) {
      int mm = i + 8 * q;
      int si = sSrc[mm], di = sDst[mm];
      float kv = QKV[(long)si * 384 + 128 + col];
      float qv = QKV[(long)di * 384 + col];
      float p  = kv * qv * accE[i];
      p += __shfl_xor(p, 1); p += __shfl_xor(p, 2);
      p += __shfl_xor(p, 4); p += __shfl_xor(p, 8);   // reduce over dh within 16-lane half
      float sc = fminf(fmaxf(p * 0.25f + sBias[mm][h], -5.0f), 5.0f);
      float sw = expf(sc);
      float vv = QKV[(long)si * 384 + 256 + col];
      atomicAdd(&wV[(long)di * 128 + col], vv * sw);
      if (r == 0) atomicAdd(&Zacc[di * 8 + h], sw);
    }
  }
}

// ---------------- BN1 + residual, emit x (f32) and xb (bf16) ----------------
__global__ void bn1_kernel(const float* __restrict__ h, const float* __restrict__ wV,
                           const float* __restrict__ Zacc,
                           const float* __restrict__ g, const float* __restrict__ b,
                           const float* __restrict__ mu, const float* __restrict__ var,
                           float* __restrict__ x, bf16t* __restrict__ xb) {
  long i = (long)blockIdx.x * 256 + threadIdx.x;
  if (i >= (long)NROWS * 128) return;
  int c = (int)(i & 127);
  long row = i >> 7;
  float attn = wV[i] / (Zacc[row * 8 + (c >> 4)] + 1e-6f);
  float xv = h[i] + attn;
  xv = (xv - mu[c]) * g[c] * rsqrtf(var[c] + 1e-5f) + b[c];
  x[i] = xv;
  xb[i] = (bf16t)xv;
}

// ---------------- FF1: y1 = relu(x @ Wff1 + b1), bf16 out ----------------
__global__ void ff1_gemm(const bf16t* __restrict__ xb, const bf16t* __restrict__ W1p,
                         const float* __restrict__ b1, bf16t* __restrict__ y1b) {
  int rt = blockIdx.x, nt = blockIdx.y;     // 16 n-tiles
  int L = threadIdx.x, q = L >> 4, r = L & 15;
  const bf16t* Arow = xb + (long)(rt * 16 + r) * 128;
  v8f acc = {};
#pragma unroll
  for (int s = 0; s < 4; ++s)
    acc = wmma_bf16(load_afrag(Arow + 32 * s + 8 * q),
                    load_bfrag(W1p + ((s * 16 + nt) << 9) + L * 16), acc);
  int n = nt * 16 + r;
  float bias = b1[n];
#pragma unroll
  for (int i = 0; i < 8; ++i)
    y1b[(long)(rt * 16 + i + 8 * q) * 256 + n] = (bf16t)fmaxf(acc[i] + bias, 0.0f);
}

// ---------------- FF2 + BN2 epilogue -> out ----------------
__global__ void ff2_gemm(const bf16t* __restrict__ y1b, const bf16t* __restrict__ W2p,
                         const float* __restrict__ b2, const float* __restrict__ x,
                         const float* __restrict__ g, const float* __restrict__ bb,
                         const float* __restrict__ mu, const float* __restrict__ var,
                         float* __restrict__ out) {
  int rt = blockIdx.x, nt = blockIdx.y;     // 8 n-tiles, K=256 -> 8 k-steps
  int L = threadIdx.x, q = L >> 4, r = L & 15;
  const bf16t* Arow = y1b + (long)(rt * 16 + r) * 256;
  v8f acc = {};
#pragma unroll
  for (int s = 0; s < 8; ++s)
    acc = wmma_bf16(load_afrag(Arow + 32 * s + 8 * q),
                    load_bfrag(W2p + ((s * 8 + nt) << 9) + L * 16), acc);
  int n = nt * 16 + r;
  float bias = b2[n];
  float gam = g[n], bet = bb[n], m2 = mu[n], inv = rsqrtf(var[n] + 1e-5f);
#pragma unroll
  for (int i = 0; i < 8; ++i) {
    long row = (long)(rt * 16 + i + 8 * q);
    float ff = acc[i] + bias;
    out[row * 128 + n] = (x[row * 128 + n] + ff - m2) * gam * inv + bet;
  }
}

// ---------------- host launch ----------------
extern "C" void kernel_launch(void* const* d_in, const int* in_sizes, int n_in,
                              void* d_out, int out_size, void* d_ws, size_t ws_size,
                              hipStream_t stream) {
  const float*     h_in  = (const float*)d_in[0];
  const long long* eidx  = (const long long*)d_in[1];
  const float*     eattr = (const float*)d_in[2];
  const float *WQ = (const float*)d_in[3], *WK = (const float*)d_in[4],
              *WV = (const float*)d_in[5], *WE = (const float*)d_in[6],
              *WEb = (const float*)d_in[7], *bEb = (const float*)d_in[8],
              *Wff1 = (const float*)d_in[9], *bff1 = (const float*)d_in[10],
              *Wff2 = (const float*)d_in[11], *bff2 = (const float*)d_in[12];
  const float *g1 = (const float*)d_in[13], *b1 = (const float*)d_in[14],
              *m1 = (const float*)d_in[15], *v1 = (const float*)d_in[16];
  const float *g2 = (const float*)d_in[17], *b2 = (const float*)d_in[18],
              *m2 = (const float*)d_in[19], *v2 = (const float*)d_in[20];
  float* out = (float*)d_out;

  char* ws = (char*)d_ws;
  auto alloc = [&](size_t bytes) {
    char* p = ws; ws += (bytes + 255) & ~(size_t)255; return p;
  };
  bf16t* hb    = (bf16t*)alloc((size_t)NROWS * 128 * 2);
  float* QKV   = (float*)alloc((size_t)NROWS * 384 * 4);
  float* wV    = (float*)alloc((size_t)NROWS * 128 * 4);
  float* Zacc  = (float*)alloc((size_t)NROWS * 8 * 4);
  float* x     = (float*)alloc((size_t)NROWS * 128 * 4);
  bf16t* xb    = (bf16t*)alloc((size_t)NROWS * 128 * 2);
  bf16t* y1b   = (bf16t*)alloc((size_t)NROWS * 256 * 2);
  bf16t* WQKVp = (bf16t*)alloc((size_t)4 * 24 * 512 * 2);
  bf16t* WEp   = (bf16t*)alloc((size_t)4 * 8 * 512 * 2);
  bf16t* WEbp  = (bf16t*)alloc((size_t)4 * 1 * 512 * 2);
  bf16t* W1p   = (bf16t*)alloc((size_t)4 * 16 * 512 * 2);
  bf16t* W2p   = (bf16t*)alloc((size_t)8 * 8 * 512 * 2);

  // weight packing + h conversion + accumulator zeroing
  pack_qkv<<<(4 * 24 * 512 + 255) / 256, 256, 0, stream>>>(WQ, WK, WV, WQKVp);
  pack_w<<<(4 * 8 * 512 + 255) / 256, 256, 0, stream>>>(WE, WEp, 4, 8, 128);
  pack_w<<<(4 * 1 * 512 + 255) / 256, 256, 0, stream>>>(WEb, WEbp, 4, 1, 8);
  pack_w<<<(4 * 16 * 512 + 255) / 256, 256, 0, stream>>>(Wff1, W1p, 4, 16, 256);
  pack_w<<<(8 * 8 * 512 + 255) / 256, 256, 0, stream>>>(Wff2, W2p, 8, 8, 128);
  f32_to_bf16<<<(NROWS * 128 + 255) / 256, 256, 0, stream>>>(h_in, hb, NROWS * 128);
  zero_f32<<<(NROWS * 128 + 255) / 256, 256, 0, stream>>>(wV, NROWS * 128);
  zero_f32<<<(NROWS * 8 + 255) / 256, 256, 0, stream>>>(Zacc, NROWS * 8);

  // QKV projection (WMMA)
  qkv_gemm<<<dim3(NROWS / 16, 3), 32, 0, stream>>>(hb, WQKVp, QKV);
  // fused edge GEMM + attention scatter (WMMA + async-LDS + atomics)
  edge_kernel<<<EDGES / 16, 32, 0, stream>>>(eattr, eidx, QKV, WEp, WEbp, bEb, wV, Zacc);
  // BN1 + residual
  bn1_kernel<<<(NROWS * 128 + 255) / 256, 256, 0, stream>>>(h_in, wV, Zacc, g1, b1, m1, v1, x, xb);
  // FFN (WMMA) with fused BN2 epilogue
  ff1_gemm<<<dim3(NROWS / 16, 16), 32, 0, stream>>>(xb, W1p, bff1, y1b);
  ff2_gemm<<<dim3(NROWS / 16, 8), 32, 0, stream>>>(y1b, W2p, bff2, x, g2, b2, m2, v2, out);

  (void)in_sizes; (void)n_in; (void)out_size; (void)ws_size;
}
